// TTNetShared_78013785964679
// MI455X (gfx1250) — matrix operand order repaired and checked by reading, not compiled, not run on hardware
//
#include <hip/hip_runtime.h>
#include <hip/hip_bf16.h>

typedef __attribute__((ext_vector_type(16))) _Float16 v16h;
typedef __attribute__((ext_vector_type(8)))  float    v8f;
typedef __attribute__((ext_vector_type(4)))  float    v4f;

// half-wave swap: lane l <-> lane l^16  (group-of-32 swizzle, xor=0x10, and=0x1f)
static __device__ __forceinline__ float swz16(float x) {
  int i = __builtin_amdgcn_ds_swizzle(__builtin_bit_cast(int, x), 0x401f);
  return __builtin_bit_cast(float, i);
}

// One wave owns 16 batch rows. State (16 rows x 32) lives in WMMA B-layout f32
// registers between steps; per step we rebuild the f16 B operand (with cos/sin
// folded in as a per-lane scale) and run 4 chained v_wmma_f32_16x16x32_f16.
__global__ __launch_bounds__(256, 2) void ttnet_fwd(
    const float* __restrict__ X,   // (batch, T)
    const float* __restrict__ Wf,  // (1, 2, 32)
    const float* __restrict__ Ws,  // (32, 2, 32)  flat [i*64 + j*32 + k]
    const float* __restrict__ Wl,  // (32, 2, 10)  flat [i*20 + j*10 + n]
    float* __restrict__ out,       // (batch, 10)
    int batch, int T)
{
  const int  lane = threadIdx.x & 31;
  const int  wave = threadIdx.x >> 5;
  const int  lm   = lane & 15;
  const bool up   = lane >= 16;
  int b = blockIdx.x * 128 + wave * 16 + lm;
  const bool valid = b < batch;
  if (b >= batch) b = batch - 1;           // clamp: keep EXEC all-ones for WMMA

  // ---- A operands: A0_lo[m][K] = W0^T[m][K] = Ws[K*64 + m]  (loaded once) ----
  // A k order used here (contiguous): lane<16 holds M=lm, halves h: K = {0..15};
  //                  lane>=16 holds M=lm, K = {16..31}
  v16h a0lo, a0hi, a1lo, a1hi;
#pragma unroll
  for (int h = 0; h < 16; ++h) {
    const int K = h + (up ? 16 : 0);
    const float* w = Ws + K * 64 + lm;
    a0lo[h] = (_Float16)w[0];    // j=0, k = lm
    a0hi[h] = (_Float16)w[16];   // j=0, k = lm+16
    a1lo[h] = (_Float16)w[32];   // j=1, k = lm
    a1hi[h] = (_Float16)w[48];   // j=1, k = lm+16
  }

  // ---- initial state (B-layout, f32): state0[k] = Wf[0,0,k] + Wf[0,1,k] ----
  // lane<16: val[h] = state[k=h]; lane>=16: val[h] = state[k=h+16]
  float val[16];
#pragma unroll
  for (int h = 0; h < 16; ++h) {
    const int k = h + (up ? 16 : 0);
    val[h] = Wf[k] + Wf[32 + k];
  }

  // ---- scan over T with float4 lookahead + L2 prefetch ----
  const float* xrow = X + (size_t)b * T;
  v4f chunk = *(const v4f*)xrow;
  for (int t0 = 0; t0 < T; t0 += 4) {
    v4f next = chunk;
    if (t0 + 4 < T)  next = *(const v4f*)(xrow + t0 + 4);
    if (t0 + 32 < T) __builtin_prefetch(xrow + t0 + 32, 0, 3);  // global_prefetch_b8
#pragma unroll
    for (int u = 0; u < 4; ++u) {
      const float x = chunk[u];
      // cos(pi*x/2) = cos(2*pi * x/4): HW trig input is in revolutions
      const float c = __builtin_amdgcn_cosf(0.25f * x);
      const float s = __builtin_amdgcn_sinf(0.25f * x);
      v16h bc, bs;                       // per-lane (= per-batch-column) scaling
#pragma unroll
      for (int h = 0; h < 16; ++h) {
        bc[h] = (_Float16)(c * val[h]);
        bs[h] = (_Float16)(s * val[h]);
      }
      v8f dlo = {}, dhi = {};
      dlo = __builtin_amdgcn_wmma_f32_16x16x32_f16(false, a0lo, false, bc, (short)0, dlo, false, false);
      dlo = __builtin_amdgcn_wmma_f32_16x16x32_f16(false, a1lo, false, bs, (short)0, dlo, false, false);
      dhi = __builtin_amdgcn_wmma_f32_16x16x32_f16(false, a0hi, false, bc, (short)0, dhi, false, false);
      dhi = __builtin_amdgcn_wmma_f32_16x16x32_f16(false, a1hi, false, bs, (short)0, dhi, false, false);
      // D layout -> B layout: half-wave exchange of the "other" 8 k-values
      // lane<16 has k{0..7}=dlo, k{16..23}=dhi; lane>=16 has k{8..15}=dlo, k{24..31}=dhi
#pragma unroll
      for (int h = 0; h < 8; ++h) {
        const float send = up ? dlo[h] : dhi[h];
        const float got  = swz16(send);
        val[h]     = up ? got    : dlo[h];   // k = h      (+16 on upper half)
        val[h + 8] = up ? dhi[h] : got;      // k = h + 8  (+16 on upper half)
      }
    }
    chunk = next;
  }

  // ---- projection to 10 classes: out[n] = sum_k state[k]*(Wl[k,0,n]+Wl[k,1,n]) ----
  float p[10];
#pragma unroll
  for (int n = 0; n < 10; ++n) p[n] = 0.f;
#pragma unroll
  for (int h = 0; h < 16; ++h) {
    const int k = h + (up ? 16 : 0);
    const float* wl = Wl + k * 20;
    const float v = val[h];
#pragma unroll
    for (int n = 0; n < 10; ++n) p[n] = fmaf(v, wl[n] + wl[10 + n], p[n]);
  }
#pragma unroll
  for (int n = 0; n < 10; ++n) p[n] += swz16(p[n]);   // pair (l, l+16) reduction

  // ---- log_softmax, lower half-lane writes its batch row ----
  if (!up && valid) {
    float mx = p[0];
#pragma unroll
    for (int n = 1; n < 10; ++n) mx = fmaxf(mx, p[n]);
    float sum = 0.f;
#pragma unroll
    for (int n = 0; n < 10; ++n) sum += __expf(p[n] - mx);
    const float lse = mx + __logf(sum);
    float* o = out + (size_t)b * 10;
#pragma unroll
    for (int n = 0; n < 10; ++n) o[n] = p[n] - lse;
  }
}

extern "C" void kernel_launch(void* const* d_in, const int* in_sizes, int n_in,
                              void* d_out, int out_size, void* d_ws, size_t ws_size,
                              hipStream_t stream) {
  const float* X  = (const float*)d_in[0];   // (batch, T)
  const float* Wf = (const float*)d_in[1];   // (1,2,32)
  const float* Ws = (const float*)d_in[2];   // (32,2,32)
  const float* Wl = (const float*)d_in[3];   // (32,2,10)
  float* out = (float*)d_out;                // (batch, 10)

  const int batch = out_size / 10;
  const int T     = in_sizes[0] / batch;

  const int rows_per_block = 128;            // 8 waves x 16 batch rows
  const int grid = (batch + rows_per_block - 1) / rows_per_block;
  ttnet_fwd<<<grid, 256, 0, stream>>>(X, Wf, Ws, Wl, out, batch, T);
}
